// AgGemmTriton_35510789603645
// MI455X (gfx1250) — 8-bit matrix operand order repaired and checked by reading, compiled, not run on hardware
//
#include <hip/hip_runtime.h>
#include <hip/hip_bf16.h>
#include <stdint.h>

typedef int v8i __attribute__((ext_vector_type(8)));
typedef int v4i __attribute__((ext_vector_type(4)));
typedef int v2i __attribute__((ext_vector_type(2)));

#define M_DIM 4096
#define K_DIM 8192
#define N_DIM 8192

#define BM 128
#define BN 128
#define BK 64
#define KTILES (K_DIM / BK)

#define TILE_BYTES  (BM * BK)
#define STAGE_BYTES (2 * TILE_BYTES)

__device__ __forceinline__ void copy16_to_lds(int8_t* ldst, const int8_t* gsrc) {
#if __has_builtin(__builtin_amdgcn_global_load_async_to_lds_b128)
    __builtin_amdgcn_global_load_async_to_lds_b128(
        (__attribute__((address_space(1))) v4i*)gsrc,
        (__attribute__((address_space(3))) v4i*)ldst,
         0,  0);
#else
    *(v4i*)ldst = *(const v4i*)gsrc;
#endif
}

__device__ __forceinline__ void drain_async_copies() {
#if __has_builtin(__builtin_amdgcn_global_load_async_to_lds_b128)
#if __has_builtin(__builtin_amdgcn_s_wait_asynccnt)
    __builtin_amdgcn_s_wait_asynccnt(0);
#else
    asm volatile("s_wait_asynccnt 0" ::: "memory");
#endif
#endif
}

__global__ __launch_bounds__(256)
void i8gemm_wmma_kernel(const int8_t* __restrict__ x,
                        const int8_t* __restrict__ w,
                        const float*  __restrict__ bias,
                        const float*  __restrict__ iscale,
                        const float*  __restrict__ wscale,
                        __hip_bfloat16* __restrict__ out)
{
    __shared__ __align__(16) int8_t smem[2 * STAGE_BYTES];

    const int tid   = threadIdx.x;
    const int lane  = tid & 31;
    const int wave  = tid >> 5;
    const int waveM = wave >> 2;
    const int waveN = wave & 3;
    const int laneHalf = lane >> 4;
    const int lane15   = lane & 15;

    const int bM = blockIdx.y * BM;
    const int bN = blockIdx.x * BN;

    const int c0 = tid, c1 = tid + 256;
    const int r0 = c0 >> 2, o0 = (c0 & 3) * 16;
    const int r1 = c1 >> 2, o1 = (c1 & 3) * 16;
    const int la0 = r0 * BK + o0;
    const int la1 = r1 * BK + o1;

    const int8_t* xrow0 = x + (size_t)(bM + r0) * K_DIM + o0;
    const int8_t* xrow1 = x + (size_t)(bM + r1) * K_DIM + o1;
    const int8_t* wrow0 = w + (size_t)(bN + r0) * K_DIM + o0;
    const int8_t* wrow1 = w + (size_t)(bN + r1) * K_DIM + o1;

    const int aoff = (waveM * 64 + lane15) * BK + laneHalf * 16;
    const int boff = (waveN * 32 + lane15) * BK + laneHalf * 16;

    v8i acc[4][2];
#pragma unroll
    for (int i = 0; i < 4; ++i)
#pragma unroll
        for (int j = 0; j < 2; ++j)
            acc[i][j] = {};

    copy16_to_lds(smem + la0, xrow0);
    copy16_to_lds(smem + la1, xrow1);
    copy16_to_lds(smem + TILE_BYTES + la0, wrow0);
    copy16_to_lds(smem + TILE_BYTES + la1, wrow1);
    drain_async_copies();
    __syncthreads();

    int stage = 0;
#pragma unroll 1
    for (int kt = 0; kt < KTILES; ++kt) {
        {
            const size_t ko = (size_t)(((kt + 1) & (KTILES - 1)) * BK);
            int8_t* nxt = smem + (stage ^ 1) * STAGE_BYTES;
            copy16_to_lds(nxt + la0, xrow0 + ko);
            copy16_to_lds(nxt + la1, xrow1 + ko);
            copy16_to_lds(nxt + TILE_BYTES + la0, wrow0 + ko);
            copy16_to_lds(nxt + TILE_BYTES + la1, wrow1 + ko);
        }

        const int8_t* A = smem + stage * STAGE_BYTES;
        const int8_t* B = A + TILE_BYTES;

        v8i bfrag[2];
#pragma unroll
        for (int tn = 0; tn < 2; ++tn) {
            const int8_t* bp = B + boff + tn * 16 * BK;
            v4i lo = *(const v4i*)(bp +  0);
            v4i hi = *(const v4i*)(bp + 32);
            v8i b;
            b[0] = lo[0]; b[1] = lo[1]; b[2] = lo[2]; b[3] = lo[3];
            b[4] = hi[0]; b[5] = hi[1]; b[6] = hi[2]; b[7] = hi[3];
            bfrag[tn] = b;
        }

#pragma unroll
        for (int tm = 0; tm < 4; ++tm) {
            const int8_t* ap = A + aoff + tm * 16 * BK;
            v2i p0 = *(const v2i*)(ap +  0);
            v2i p1 = *(const v2i*)(ap +  8);
            v2i p2 = *(const v2i*)(ap + 32);
            v2i p3 = *(const v2i*)(ap + 40);
            v8i a;
            a[0] = p0[0]; a[1] = p0[1];
            a[2] = p1[0]; a[3] = p1[1];
            a[4] = p2[0]; a[5] = p2[1];
            a[6] = p3[0]; a[7] = p3[1];

            acc[tm][0] = __builtin_amdgcn_wmma_i32_16x16x64_iu8(
                true, a, true, bfrag[0], acc[tm][0], false, false);
            acc[tm][1] = __builtin_amdgcn_wmma_i32_16x16x64_iu8(
                true, a, true, bfrag[1], acc[tm][1], false, false);
        }

        drain_async_copies();
        __syncthreads();
        stage ^= 1;
    }

#pragma unroll
    for (int tm = 0; tm < 4; ++tm) {
        const int rowbase = bM + waveM * 64 + tm * 16 + laneHalf * 8;
        float is[8];
#pragma unroll
        for (int r = 0; r < 8; ++r) is[r] = iscale[rowbase + r];
#pragma unroll
        for (int tn = 0; tn < 2; ++tn) {
            const int col = bN + waveN * 32 + tn * 16 + lane15;
            const float ws = wscale[col];
            const float bs = bias[col];
#pragma unroll
            for (int r = 0; r < 8; ++r) {
                const float o = (float)acc[tm][tn][r] * is[r] * ws + bs;
                out[(size_t)(rowbase + r) * N_DIM + col] = __float2bfloat16(o);
            }
        }
    }
}

extern "C" void kernel_launch(void* const* d_in, const int* in_sizes, int n_in,
                              void* d_out, int out_size, void* d_ws, size_t ws_size,
                              hipStream_t stream) {
    const int8_t* x      = (const int8_t*)d_in[0];
    const int8_t* w      = (const int8_t*)d_in[1];
    const float*  bias   = (const float*)d_in[2];
    const float*  iscale = (const float*)d_in[3];
    const float*  wscale = (const float*)d_in[4];
    __hip_bfloat16* out  = (__hip_bfloat16*)d_out;

    dim3 grid(N_DIM / BN, M_DIM / BM);
    i8gemm_wmma_kernel<<<grid, 256, 0, stream>>>(x, w, bias, iscale, wscale, out);
}
